// SIMPLE_LayerNorm_LSTM_10333691314502
// MI455X (gfx1250) — matrix operand order repaired and checked by reading, not compiled, not run on hardware
//
#include <hip/hip_runtime.h>
#include <hip/hip_bf16.h>
#include <cstdint>

typedef _Float16 v8h  __attribute__((ext_vector_type(8)));
typedef _Float16 v16h __attribute__((ext_vector_type(16)));
typedef float    v8f  __attribute__((ext_vector_type(8)));
typedef uint32_t u32x4 __attribute__((ext_vector_type(4)));
typedef int32_t  i32x8 __attribute__((ext_vector_type(8)));

#define T_STEPS 512
#define BATCH   256
#define IDIM    256
#define HDIM    256
#define KDIM    512    // IDIM + HDIM (combined GEMM K)
#define MTILE   16     // batch rows per workgroup

__device__ __forceinline__ float wave_sum(float v) {
#pragma unroll
  for (int m = 16; m >= 1; m >>= 1) v += __shfl_xor(v, m, 32);
  return v;
}
__device__ __forceinline__ float sigmoidf(float x) {
  return 1.0f / (1.0f + __expf(-x));
}

// TDM: async-load a 16x256 fp32 tile (row stride 256 floats) into LDS.
// D# built per CDNA5 ISA; 2D short form (groups 0,1).
// Tracked by TENSORcnt; pair with __builtin_amdgcn_s_wait_tensorcnt.
__device__ __forceinline__ void tdm_load_x_tile(const float* gsrc,
                                                uint32_t lds_byte_off) {
  const uint64_t ga = (uint64_t)(uintptr_t)gsrc;
  u32x4 g0;
  g0.x = 1u;                                  // count=1 (valid), user mode
  g0.y = lds_byte_off;                        // lds_addr (bytes)
  g0.z = (uint32_t)(ga & 0xffffffffu);        // global_addr[31:0]
  g0.w = (uint32_t)((ga >> 32) & 0x01ffffffu) // global_addr[56:32]
         | (2u << 30);                        // type = 2 ("image")
  i32x8 g1;
  g1[0] = (int)(2u << 16);       // data_size = 2 (4 bytes); no pad/iter/mask
  g1[1] = (int)(256u << 16);     // tensor_dim0[15:0] = 256 (bits 63:48)
  g1[2] = (int)(16u << 16);      // tensor_dim0[31:16]=0 | tensor_dim1[15:0]=16
  g1[3] = (int)(256u << 16);     // tensor_dim1[31:16]=0 | tile_dim0 = 256
  g1[4] = 16;                    // tile_dim1 = 16 | tile_dim2 = 0
  g1[5] = 256;                   // tensor_dim0_stride[31:0] = 256 elements
  g1[6] = 0;                     // stride[47:32]=0 | tensor_dim1_stride[15:0]=0
  g1[7] = 0;
  asm volatile("tensor_load_to_lds %0, %1" :: "s"(g0), "s"(g1) : "memory");
}

// Build combined f16 weight matrix Wc[n][k], n in [0,1024), k in [0,512):
//   k <  256 : W_ih[n][k]      k >= 256 : W_hh[n][k-256]
// Row-major with K contiguous => direct v16h B-fragment loads for WMMA.
__global__ void prep_weights(const float* __restrict__ Wih,
                             const float* __restrict__ Whh,
                             _Float16* __restrict__ Wc) {
  const int n = blockIdx.x;
  for (int k = threadIdx.x; k < KDIM; k += blockDim.x) {
    float v = (k < IDIM) ? Wih[n * IDIM + k] : Whh[n * HDIM + (k - IDIM)];
    Wc[(size_t)n * KDIM + k] = (_Float16)v;
  }
}

// Persistent LayerNorm-LSTM: 16 workgroups x 16 batch rows, 512 threads
// (16 waves). Wave w owns gate columns [w*16, w*16+16) of i/f/g/o so the
// whole cell update is wave-local in the WMMA accumulators. Next step's
// x-tile is TDM-prefetched into a double-buffered LDS stage.
__launch_bounds__(512, 1)
__global__ void lnlstm_kernel(const float* __restrict__ x,
                              const float* __restrict__ hx,
                              const float* __restrict__ cx,
                              const _Float16* __restrict__ Wc,
                              const float* __restrict__ bih,
                              const float* __restrict__ bhh,
                              const float* __restrict__ gh,
                              const float* __restrict__ beh,
                              const float* __restrict__ gc,
                              const float* __restrict__ bec,
                              float* __restrict__ out) {
  __shared__ _Float16 A_lds[MTILE][KDIM];        // [x_t | LN(h)] f16 (A op)
  __shared__ float    h_lds[MTILE][HDIM];        // recurrent h (fp32)
  __shared__ float    c_lds[MTILE][HDIM];        // recurrent c (fp32)
  __shared__ float    cn_lds[MTILE][HDIM];       // LN(c) for the cell update
  __shared__ float    x_stage[2][MTILE][IDIM];   // TDM double-buffer for x_t

  const int tid  = threadIdx.x;
  const int wave = tid >> 5;
  const int lane = tid & 31;
  const int lh   = lane & 15;   // WMMA: A row / B column / D column
  const int hi   = lane >> 4;   // WMMA: lane-half (K-half select, D row-half)
  const int b0   = blockIdx.x * MTILE;

  const uint32_t xs_off0 = (uint32_t)(uintptr_t)(&x_stage[0][0][0]);
  const uint32_t xs_off1 = (uint32_t)(uintptr_t)(&x_stage[1][0][0]);

  // Phase-1 constants (wave = batch row, lane covers 8 columns)
  float ghv[8], behv[8], gcv[8], becv[8];
#pragma unroll
  for (int i = 0; i < 8; ++i) {
    const int j = lane * 8 + i;
    ghv[i] = gh[j]; behv[i] = beh[j];
    gcv[i] = gc[j]; becv[i] = bec[j];
  }

  // Phase-3 constants: wave w, lane handles gate column jc for rows 8*hi..8*hi+7.
  const int jc = wave * 16 + lh;
  float bb[4];
  const _Float16* Bbase[4];
#pragma unroll
  for (int g = 0; g < 4; ++g) {
    bb[g]    = bih[g * HDIM + jc] + bhh[g * HDIM + jc];
    Bbase[g] = Wc + (size_t)(g * HDIM + jc) * KDIM;
  }

  // Kick off TDM for the t=0 x-tile while we initialize h, c.
  if (wave == 0) tdm_load_x_tile(x + (size_t)b0 * IDIM, xs_off0);
  {
    const int row = wave;
#pragma unroll
    for (int i = 0; i < 8; ++i) {
      const int j = lane * 8 + i;
      h_lds[row][j] = hx[(b0 + row) * HDIM + j];
      c_lds[row][j] = cx[(b0 + row) * HDIM + j];
    }
  }
  if (wave == 0) __builtin_amdgcn_s_wait_tensorcnt(0);
  __syncthreads();

  for (int t = 0; t < T_STEPS; ++t) {
    // Prefetch x_{t+1} into the other stage buffer (off the critical path).
    if (wave == 0 && (t + 1) < T_STEPS) {
      tdm_load_x_tile(x + ((size_t)(t + 1) * BATCH + (size_t)b0) * IDIM,
                      ((t + 1) & 1) ? xs_off1 : xs_off0);
    }

    // ---- Phase 1: LN(h), LN(c), stage x_t as f16 (wave = row) ----
    {
      const int row = wave;
      float hv[8], cv[8];
      float sh = 0.f, shh = 0.f, sc = 0.f, scc = 0.f;
#pragma unroll
      for (int i = 0; i < 8; ++i) {
        const int j = lane * 8 + i;
        const float h = h_lds[row][j], c = c_lds[row][j];
        hv[i] = h; cv[i] = c;
        sh += h; shh += h * h;
        sc += c; scc += c * c;
      }
      sh = wave_sum(sh); shh = wave_sum(shh);
      sc = wave_sum(sc); scc = wave_sum(scc);
      const float inv = 1.0f / (float)HDIM;
      const float muh = sh * inv, varh = shh * inv - muh * muh;
      const float muc = sc * inv, varc = scc * inv - muc * muc;
      const float rsh = rsqrtf(varh + 1e-5f);
      const float rsc = rsqrtf(varc + 1e-5f);
      const float* xr = &x_stage[t & 1][row][0];  // TDM-staged tile (LDS)
#pragma unroll
      for (int i = 0; i < 8; ++i) {
        const int j  = lane * 8 + i;
        const float hn = (hv[i] - muh) * rsh * ghv[i] + behv[i];
        const float cn = (cv[i] - muc) * rsc * gcv[i] + becv[i];
        A_lds[row][IDIM + j] = (_Float16)hn;
        cn_lds[row][j]       = cn;
        A_lds[row][j]        = (_Float16)xr[j];
      }
    }
    __syncthreads();

    // ---- Phase 2: gates = [x|h_n] @ Wc^T via v_wmma_f32_16x16x32_f16 ----
    v8f acc0 = {}, acc1 = {}, acc2 = {}, acc3 = {};
#pragma unroll 4
    for (int k0 = 0; k0 < KDIM; k0 += 32) {
      // A fragment: lane half hi -> K range k0+hi*16..+15 (contiguous)
      const _Float16* pa = &A_lds[lh][k0 + hi * 16];
      const v8h alo = *(const v8h*)pa;
      const v8h ahi = *(const v8h*)(pa + 8);
      const v16h af = __builtin_shufflevector(
          alo, ahi, 0, 1, 2, 3, 4, 5, 6, 7, 8, 9, 10, 11, 12, 13, 14, 15);
      // B fragments: lane lh = column, hi selects K-half, 16 contiguous halves
      const v16h bf0 = *(const v16h*)(Bbase[0] + k0 + hi * 16);
      const v16h bf1 = *(const v16h*)(Bbase[1] + k0 + hi * 16);
      const v16h bf2 = *(const v16h*)(Bbase[2] + k0 + hi * 16);
      const v16h bf3 = *(const v16h*)(Bbase[3] + k0 + hi * 16);
      acc0 = __builtin_amdgcn_wmma_f32_16x16x32_f16(false, af, false, bf0,
                                                    (short)0, acc0, false, false);
      acc1 = __builtin_amdgcn_wmma_f32_16x16x32_f16(false, af, false, bf1,
                                                    (short)0, acc1, false, false);
      acc2 = __builtin_amdgcn_wmma_f32_16x16x32_f16(false, af, false, bf2,
                                                    (short)0, acc2, false, false);
      acc3 = __builtin_amdgcn_wmma_f32_16x16x32_f16(false, af, false, bf3,
                                                    (short)0, acc3, false, false);
    }

    // ---- Phase 3: wave-local cell update; acc[g][r] is gate g at (m, jc) ----
#pragma unroll
    for (int r = 0; r < 8; ++r) {
      const int m = r + hi * 8;  // D layout: lanes 16-31 hold rows 8-15
      const float ig = sigmoidf(acc0[r] + bb[0]);
      const float fg = sigmoidf(acc1[r] + bb[1]);
      const float gg = tanhf(acc2[r] + bb[2]);
      const float og = sigmoidf(acc3[r] + bb[3]);
      const float cnew = fg * cn_lds[m][jc] + ig * gg;
      const float hnew = og * tanhf(cnew);
      h_lds[m][jc] = hnew;
      c_lds[m][jc] = cnew;
      out[(size_t)t * (BATCH * HDIM) + (size_t)(b0 + m) * HDIM + jc] =
          fmaxf(hnew, 0.0f);
    }
    // Ensure the prefetched tile has landed before anyone reads it next step.
    if (wave == 0) __builtin_amdgcn_s_wait_tensorcnt(0);
    __syncthreads();
  }

  // ---- Final carry: (out, h_f, c_f) concatenated flat ----
  {
    const int row = wave;
    float* hout = out + (size_t)T_STEPS * BATCH * HDIM;
    float* cout = hout + (size_t)BATCH * HDIM;
#pragma unroll
    for (int i = 0; i < 8; ++i) {
      const int j = lane * 8 + i;
      hout[(size_t)(b0 + row) * HDIM + j] = h_lds[row][j];
      cout[(size_t)(b0 + row) * HDIM + j] = c_lds[row][j];
    }
  }
}

extern "C" void kernel_launch(void* const* d_in, const int* in_sizes, int n_in,
                              void* d_out, int out_size, void* d_ws,
                              size_t ws_size, hipStream_t stream) {
  const float* input = (const float*)d_in[0];
  const float* hx    = (const float*)d_in[1];
  const float* cx    = (const float*)d_in[2];
  const float* Wih   = (const float*)d_in[3];
  const float* Whh   = (const float*)d_in[4];
  const float* bih   = (const float*)d_in[5];
  const float* bhh   = (const float*)d_in[6];
  const float* gh    = (const float*)d_in[7];
  const float* beh   = (const float*)d_in[8];
  const float* gc    = (const float*)d_in[9];
  const float* bec   = (const float*)d_in[10];
  float* out = (float*)d_out;

  _Float16* Wc = (_Float16*)d_ws;  // 1024*512*2 = 1 MB

  prep_weights<<<4 * HDIM, 256, 0, stream>>>(Wih, Whh, Wc);
  lnlstm_kernel<<<BATCH / MTILE, 512, 0, stream>>>(
      input, hx, cx, Wc, bih, bhh, gh, beh, gc, bec, out);
}
